// ASTGCNBlock_21715354649721
// MI455X (gfx1250) — compile-verified, matrix operand order checked by reading, not run on hardware
//
#include <hip/hip_runtime.h>
#include <math.h>

// ---------------- problem constants (reference shapes) ----------------
#define BB   4
#define NN   2048
#define FIN  32
#define TT   16
#define CC   64
#define TFC  64
#define EE   32768
#define LN_EPS 1e-5f

typedef __attribute__((ext_vector_type(16))) _Float16 v16h;
typedef __attribute__((ext_vector_type(8)))  float    v8f;

__device__ __forceinline__ float sigmoidf_(float x) { return 1.0f / (1.0f + expf(-x)); }

// Per-lane fragment load for 16x32 f16 matrix whose K dimension is contiguous
// in memory per row (row stride = ld halves). Matches CDNA5 16-bit A/B layout:
// lane L holds row/col (L&15); lanes 0-15: K = 0..7 & 16..23, lanes 16-31:
// K = 8..15 & 24..31 (CDNA5 ISA 7.12.2).
__device__ __forceinline__ v16h load_frag_km(const _Float16* p, int ld) {
  int lane = threadIdx.x & 31;
  const _Float16* q = p + (size_t)(lane & 15) * ld + ((lane & 16) ? 8 : 0);
  v16h r;
#pragma unroll
  for (int i = 0; i < 8; ++i) r[i] = q[i];
#pragma unroll
  for (int i = 0; i < 8; ++i) r[8 + i] = q[16 + i];
  return r;
}

// ---------------- generic helpers ----------------
__global__ void k_zero(float* p, int n) {
  int i = blockIdx.x * blockDim.x + threadIdx.x;
  if (i < n) p[i] = 0.0f;
}
__global__ void k_f32_to_f16(const float* s, _Float16* d, int n) {
  int i = blockIdx.x * blockDim.x + threadIdx.x;
  if (i < n) d[i] = (_Float16)s[i];
}

// ---------------- stage 1: temporal attention ----------------
// rhs[b,n,t] = sum_f U3[f] * X[b,n,f,t]
__global__ void k_rhs(const float* X, const float* U3, float* rhs) {
  int i = blockIdx.x * blockDim.x + threadIdx.x;       // B*N*T
  if (i >= BB * NN * TT) return;
  int t = i % TT, bn = i / TT;
  const float* x = X + (size_t)bn * FIN * TT + t;
  float a = 0.f;
#pragma unroll
  for (int f = 0; f < FIN; ++f) a += U3[f] * x[f * TT];
  rhs[i] = a;
}
// lhs0[b,t,f] = sum_n X[b,n,f,t] * U1[n]
__global__ void k_lhs0(const float* X, const float* U1, float* lhs0) {
  int i = blockIdx.x * blockDim.x + threadIdx.x;       // B*T*FIN = 2048
  if (i >= BB * TT * FIN) return;
  int f = i % FIN, t = (i / FIN) % TT, b = i / (FIN * TT);
  float a = 0.f;
  for (int n = 0; n < NN; ++n)
    a += X[(((size_t)b * NN + n) * FIN + f) * TT + t] * U1[n];
  lhs0[(b * TT + t) * FIN + f] = a;
}
// lhs[b,t,n] = sum_f lhs0[b,t,f] * U2[f,n]
__global__ void k_lhs(const float* lhs0, const float* U2, float* lhs) {
  int i = blockIdx.x * blockDim.x + threadIdx.x;       // B*T*N
  if (i >= BB * TT * NN) return;
  int n = i % NN, t = (i / NN) % TT, b = i / (NN * TT);
  const float* l0 = lhs0 + (b * TT + t) * FIN;
  float a = 0.f;
#pragma unroll
  for (int f = 0; f < FIN; ++f) a += l0[f] * U2[f * NN + n];
  lhs[i] = a;
}
// P = lhs @ rhs ; Q = sigmoid(P+be) ; E = Ve@Q ; softmax over axis=1 (t)
__global__ void k_emat(const float* lhs, const float* rhs, const float* be,
                       const float* Ve, float* EmatG) {
  __shared__ float sP[256], sE[256];
  int b = blockIdx.x;
  int t = threadIdx.x >> 4, s = threadIdx.x & 15;
  const float* L = lhs + ((size_t)b * TT + t) * NN;
  const float* R = rhs + (size_t)b * NN * TT + s;
  float a = 0.f;
  for (int n = 0; n < NN; ++n) a += L[n] * R[n * TT];
  sP[t * 16 + s] = sigmoidf_(a + be[t * 16 + s]);
  __syncthreads();
  float e = 0.f;
#pragma unroll
  for (int u = 0; u < 16; ++u) e += Ve[t * 16 + u] * sP[u * 16 + s];
  sE[t * 16 + s] = e;
  __syncthreads();
  float mx = -3.0e38f;
#pragma unroll
  for (int u = 0; u < 16; ++u) mx = fmaxf(mx, sE[u * 16 + s]);
  float sum = 0.f;
#pragma unroll
  for (int u = 0; u < 16; ++u) sum += expf(sE[u * 16 + s] - mx);
  EmatG[(b * TT + t) * TT + s] = expf(e - mx) / sum;
}
// Xt[b,n,f,s] = sum_t X[b,n,f,t] * Emat[b,t,s]
__global__ void k_xt(const float* X, const float* EmatG, float* Xt) {
  int i = blockIdx.x * blockDim.x + threadIdx.x;       // B*N*FIN*T
  if (i >= BB * NN * FIN * TT) return;
  int s = i % TT;
  int bnf = i / TT;
  int b = bnf / (NN * FIN);
  const float* x = X + (size_t)bnf * TT;
  const float* e = EmatG + b * TT * TT + s;
  float a = 0.f;
#pragma unroll
  for (int t = 0; t < TT; ++t) a += x[t] * e[t * TT];
  Xt[i] = a;
}

// ---------------- stage 2: spatial attention ----------------
__global__ void k_ls0(const float* Xt, const float* W1, float* ls0) {
  int i = blockIdx.x * blockDim.x + threadIdx.x;       // B*N*FIN
  if (i >= BB * NN * FIN) return;
  const float* x = Xt + (size_t)i * TT;
  float a = 0.f;
#pragma unroll
  for (int t = 0; t < TT; ++t) a += x[t] * W1[t];
  ls0[i] = a;
}
__global__ void k_lsrs(const float* Xt, const float* ls0, const float* W2,
                       const float* W3, float* lhs_s, float* rhsT) {
  int i = blockIdx.x * blockDim.x + threadIdx.x;       // B*N*T
  if (i >= BB * NN * TT) return;
  int t = i % TT, bn = i / TT;
  float a = 0.f, r = 0.f;
#pragma unroll
  for (int f = 0; f < FIN; ++f) {
    a += ls0[bn * FIN + f] * W2[f * TT + t];
    r += W3[f] * Xt[(size_t)bn * FIN * TT + f * TT + t];
  }
  lhs_s[i] = a;
  rhsT[i] = r;
}
// QsT16[b,k,m] = f16( sigmoid( sum_t lhs_s[b,m,t]*rhsT[b,k,t] + bs[m,k] ) )
__global__ void k_qst(const float* lhs_s, const float* rhsT, const float* bs,
                      _Float16* QsT) {
  int i = blockIdx.x * blockDim.x + threadIdx.x;       // B*N*N
  int m = i % NN, kc = (i / NN) % NN, b = i / (NN * NN);
  const float* L = lhs_s + ((size_t)b * NN + m) * TT;
  const float* R = rhsT + ((size_t)b * NN + kc) * TT;
  float a = 0.f;
#pragma unroll
  for (int t = 0; t < TT; ++t) a += L[t] * R[t];
  QsT[i] = (_Float16)sigmoidf_(a + bs[(size_t)m * NN + kc]);
}

// Big GEMM (WMMA): ST[b,k,n] = sum_m Qs[b,m,k] * Vs[n,m]
//  A = QsT16[b] (row k, contiguous m), B = Vs16 (col n, contiguous m).
// Register-blocked: one wave computes a 32(k) x 64(n) macro-tile = 2x4 WMMA
// tiles; 6 fragment loads feed 8 WMMAs per K-step (~2.7x less L2 traffic
// than 1-tile-per-wave).
__global__ void k_gemm_wmma(const _Float16* QsT, const _Float16* Vs16, float* ST) {
  int w = (blockIdx.x * blockDim.x + threadIdx.x) >> 5;  // wave id: BB*64*32
  int tn = w & 31;          // macro col (64 wide)
  int tk = (w >> 5) & 63;   // macro row (32 tall)
  int b  = w >> 11;
  const _Float16* A0 = QsT + (size_t)b * NN * NN + (size_t)tk * 32 * NN;
  const _Float16* B0 = Vs16 + (size_t)tn * 64 * NN;
  v8f acc[2][4] = {{{}, {}, {}, {}}, {{}, {}, {}, {}}};
  for (int kk = 0; kk < NN; kk += 32) {
    __builtin_prefetch(A0 + kk + 128, 0, 0);
    __builtin_prefetch(B0 + kk + 128, 0, 0);
    v16h a[2], bf[4];
#pragma unroll
    for (int i = 0; i < 2; ++i)
      a[i] = load_frag_km(A0 + (size_t)i * 16 * NN + kk, NN);
#pragma unroll
    for (int j = 0; j < 4; ++j)
      bf[j] = load_frag_km(B0 + (size_t)j * 16 * NN + kk, NN);
#pragma unroll
    for (int i = 0; i < 2; ++i)
#pragma unroll
      for (int j = 0; j < 4; ++j)
        acc[i][j] = __builtin_amdgcn_wmma_f32_16x16x32_f16(
            false, a[i], false, bf[j], (short)0, acc[i][j], false, false);
  }
  int lane = threadIdx.x & 31;
  int moff = (lane & 16) ? 8 : 0;
#pragma unroll
  for (int i = 0; i < 2; ++i) {
    int rowbase = tk * 32 + i * 16 + moff;
#pragma unroll
    for (int j = 0; j < 4; ++j) {
      int nc = tn * 64 + j * 16 + (lane & 15);
      float* out = ST + (size_t)b * NN * NN + (size_t)rowbase * NN + nc;
#pragma unroll
      for (int v = 0; v < 8; ++v) out[(size_t)v * NN] = acc[i][j][v];
    }
  }
}

// softmax over rows of ST (= softmax over n for each (b,k), axis=1 of S)
__global__ void k_softmax_rows(float* ST) {
  __shared__ float red[256];
  float* p = ST + (size_t)blockIdx.x * NN;
  int tid = threadIdx.x;
  float mx = -3.0e38f;
  for (int n = tid; n < NN; n += 256) mx = fmaxf(mx, p[n]);
  red[tid] = mx; __syncthreads();
  for (int s = 128; s > 0; s >>= 1) { if (tid < s) red[tid] = fmaxf(red[tid], red[tid + s]); __syncthreads(); }
  mx = red[0]; __syncthreads();
  float sum = 0.f;
  for (int n = tid; n < NN; n += 256) sum += expf(p[n] - mx);
  red[tid] = sum; __syncthreads();
  for (int s = 128; s > 0; s >>= 1) { if (tid < s) red[tid] += red[tid + s]; __syncthreads(); }
  float inv = 1.0f / red[0];
  for (int n = tid; n < NN; n += 256) p[n] = expf(p[n] - mx) * inv;
}

// ---------------- stage 3: Chebyshev graph conv ----------------
__global__ void k_deg(const int* ei, float* deg) {
  int e = blockIdx.x * blockDim.x + threadIdx.x;
  if (e >= EE) return;
  int r = ei[e], c = ei[EE + e];
  if (r != c) atomicAdd(&deg[r], 1.0f);
}
__global__ void k_dinv(const float* deg, float* dinv) {
  int n = blockIdx.x * blockDim.x + threadIdx.x;
  if (n < NN) dinv[n] = deg[n] > 0.f ? rsqrtf(fmaxf(deg[n], 1e-12f)) : 0.f;
}
// att[b,e] = -dinv[r]*w*dinv[c] * S[b,r,c]  (S[b,r,c] = ST[b,c,r])
__global__ void k_att(const int* ei, const float* dinv, const float* ST, float* att) {
  int i = blockIdx.x * blockDim.x + threadIdx.x;       // B*E
  if (i >= BB * EE) return;
  int e = i % EE, b = i / EE;
  int r = ei[e], c = ei[EE + e];
  float w = (r == c) ? 0.f : 1.f;
  att[i] = -dinv[r] * w * dinv[c] * ST[((size_t)b * NN + c) * NN + r];
}
// Tx0 = X * diagS  (diagS[b,n] = ST[b,n,n])
__global__ void k_diag_tx0(const float* X, const float* ST, float* Tx0) {
  int i = blockIdx.x * blockDim.x + threadIdx.x;       // B*N*FIN*T
  if (i >= BB * NN * FIN * TT) return;
  int n = (i / (FIN * TT)) % NN, b = i / (FIN * TT * NN);
  Tx0[i] = X[i] * ST[((size_t)b * NN + n) * NN + n];
}
// dst[b,row,j] += att[b,e] * src[b,col,j]
__global__ void k_prop(const int* ei, const float* att, const float* src, float* dst) {
  int i = blockIdx.x * blockDim.x + threadIdx.x;       // B*E*(FIN*T) = 67M
  int j = i % (FIN * TT);
  int e = (i / (FIN * TT)) % EE;
  int b = i / ((FIN * TT) * EE);
  int r = ei[e], c = ei[EE + e];
  float a = att[b * EE + e];
  if (a != 0.f)
    atomicAdd(&dst[((size_t)b * NN + r) * (FIN * TT) + j],
              a * src[((size_t)b * NN + c) * (FIN * TT) + j]);
}
__global__ void k_tx2fix(float* Tx2, const float* Tx0, int n) {
  int i = blockIdx.x * blockDim.x + threadIdx.x;
  if (i < n) Tx2[i] = 2.0f * Tx2[i] - Tx0[i];
}

// Xhat[b,n,c,t] = relu( sum_k sum_f Txk[b,n,f,t]*chebW[k,f,c] + chebB[c] )
// WMMA per (b,n): M=c-tile(16), N=t(16), K=Fin(32). 12 WMMAs/wave.
__global__ void k_cheb_wmma(const float* Tx0, const float* Tx1, const float* Tx2,
                            const float* chebW, const float* chebB, float* Xhat) {
  int w = (blockIdx.x * blockDim.x + threadIdx.x) >> 5;  // B*N waves
  int n = w % NN, b = w / NN;
  int lane = threadIdx.x & 31;
  int t = lane & 15;
  int kb = (lane & 16) ? 8 : 0;
  v8f acc[4] = {{}, {}, {}, {}};
#pragma unroll
  for (int k = 0; k < 3; ++k) {
    const float* tx = (k == 0 ? Tx0 : (k == 1 ? Tx1 : Tx2)) +
                      ((size_t)b * NN + n) * (FIN * TT);
    v16h bf;  // B element (f, t) = tx[f*TT + t]
#pragma unroll
    for (int i = 0; i < 8; ++i) bf[i] = (_Float16)tx[(kb + i) * TT + t];
#pragma unroll
    for (int i = 0; i < 8; ++i) bf[8 + i] = (_Float16)tx[(kb + 16 + i) * TT + t];
    const float* Wk = chebW + k * FIN * CC;
#pragma unroll
    for (int ct = 0; ct < 4; ++ct) {
      v16h af;  // A element (m=c_local, f) = Wk[f*CC + ct*16 + m]
      int m = lane & 15;
#pragma unroll
      for (int i = 0; i < 8; ++i) af[i] = (_Float16)Wk[(kb + i) * CC + ct * 16 + m];
#pragma unroll
      for (int i = 0; i < 8; ++i) af[8 + i] = (_Float16)Wk[(kb + 16 + i) * CC + ct * 16 + m];
      acc[ct] = __builtin_amdgcn_wmma_f32_16x16x32_f16(false, af, false, bf,
                                                       (short)0, acc[ct], false, false);
    }
  }
  int moff = (lane & 16) ? 8 : 0;
  float* out = Xhat + ((size_t)b * NN + n) * (CC * TT);
#pragma unroll
  for (int ct = 0; ct < 4; ++ct)
#pragma unroll
    for (int v = 0; v < 8; ++v) {
      int c = ct * 16 + moff + v;
      out[c * TT + t] = fmaxf(acc[ct][v] + chebB[c], 0.f);
    }
}

// ---------------- stage 4: temporal conv + residual + ReLU + LayerNorm ----------------
__global__ void k_tconv_ln(const float* Xhat, const float* X,
                           const float* Wt, const float* bt,
                           const float* Wr, const float* br,
                           const float* gamma, const float* beta, float* out) {
  __shared__ float sh[256];
  int grp = blockIdx.x * 4 + (threadIdx.x >> 6);        // over B*N*T
  int o = threadIdx.x & 63;
  int t = grp % TT, n = (grp / TT) % NN, b = grp / (TT * NN);
  const float* xh = Xhat + ((size_t)b * NN + n) * (CC * TT);
  float ht = bt[o];
  for (int c = 0; c < CC; ++c) {
    const float* wt = Wt + (o * CC + c) * 3;
    const float* xr = xh + c * TT;
    float xm1 = (t > 0) ? xr[t - 1] : 0.f;
    float xp1 = (t < TT - 1) ? xr[t + 1] : 0.f;
    ht += wt[0] * xm1 + wt[1] * xr[t] + wt[2] * xp1;
  }
  const float* xx = X + ((size_t)b * NN + n) * (FIN * TT) + t;
  float hr = br[o];
#pragma unroll
  for (int f = 0; f < FIN; ++f) hr += Wr[o * FIN + f] * xx[f * TT];
  float h = fmaxf(hr + ht, 0.f);
  sh[threadIdx.x] = h;
  __syncthreads();
  int base = threadIdx.x & ~63;
  float mu = 0.f;
  for (int i = 0; i < 64; ++i) mu += sh[base + i];
  mu *= (1.f / 64.f);
  float var = 0.f;
  for (int i = 0; i < 64; ++i) { float d = sh[base + i] - mu; var += d * d; }
  var *= (1.f / 64.f);
  float y = (h - mu) * rsqrtf(var + LN_EPS) * gamma[o] + beta[o];
  out[(((size_t)b * NN + n) * TFC + o) * TT + t] = y;
}

// ---------------- host side ----------------
extern "C" void kernel_launch(void* const* d_in, const int* in_sizes, int n_in,
                              void* d_out, int out_size, void* d_ws, size_t ws_size,
                              hipStream_t stream) {
  const float* X     = (const float*)d_in[0];
  const int*   ei    = (const int*)d_in[1];
  const float* U1    = (const float*)d_in[2];
  const float* U2    = (const float*)d_in[3];
  const float* U3    = (const float*)d_in[4];
  const float* be    = (const float*)d_in[5];
  const float* Ve    = (const float*)d_in[6];
  const float* W1    = (const float*)d_in[7];
  const float* W2    = (const float*)d_in[8];
  const float* W3    = (const float*)d_in[9];
  const float* bs    = (const float*)d_in[10];
  const float* Vs    = (const float*)d_in[11];
  const float* chebW = (const float*)d_in[12];
  const float* chebB = (const float*)d_in[13];
  const float* Wt    = (const float*)d_in[14];
  const float* bt    = (const float*)d_in[15];
  const float* Wr    = (const float*)d_in[16];
  const float* br    = (const float*)d_in[17];
  const float* gamma = (const float*)d_in[18];
  const float* beta  = (const float*)d_in[19];
  float* out = (float*)d_out;

  char* ws = (char*)d_ws;
  size_t off = 0;
  auto alloc = [&](size_t bytes) -> void* {
    off = (off + 255) & ~(size_t)255;
    void* p = ws + off;
    off += bytes;
    return p;
  };
  float* rhs    = (float*)alloc((size_t)BB * NN * TT * 4);
  float* lhs0   = (float*)alloc((size_t)BB * TT * FIN * 4);
  float* lhs    = (float*)alloc((size_t)BB * TT * NN * 4);
  float* EmatG  = (float*)alloc((size_t)BB * TT * TT * 4);
  float* Xt     = (float*)alloc((size_t)BB * NN * FIN * TT * 4);
  float* ls0    = (float*)alloc((size_t)BB * NN * FIN * 4);
  float* lhs_s  = (float*)alloc((size_t)BB * NN * TT * 4);
  float* rhsT   = (float*)alloc((size_t)BB * NN * TT * 4);
  _Float16* Vs16 = (_Float16*)alloc((size_t)NN * NN * 2);
  _Float16* QsT  = (_Float16*)alloc((size_t)BB * NN * NN * 2);
  float* ST     = (float*)alloc((size_t)BB * NN * NN * 4);
  float* deg    = (float*)alloc((size_t)NN * 4);
  float* dinv   = (float*)alloc((size_t)NN * 4);
  float* att    = (float*)alloc((size_t)BB * EE * 4);
  float* Tx0    = (float*)alloc((size_t)BB * NN * FIN * TT * 4);
  float* Tx1    = (float*)alloc((size_t)BB * NN * FIN * TT * 4);
  float* Tx2    = (float*)alloc((size_t)BB * NN * FIN * TT * 4);
  float* Xhat   = (float*)alloc((size_t)BB * NN * CC * TT * 4);

  const int TPB = 256;
  const int nBNT  = BB * NN * TT;                 // 131072
  const int nBNFT = BB * NN * FIN * TT;           // 4194304

  // graph degree / normalization (independent of attention stages)
  k_zero<<<(NN + TPB - 1) / TPB, TPB, 0, stream>>>(deg, NN);
  k_deg<<<(EE + TPB - 1) / TPB, TPB, 0, stream>>>(ei, deg);
  k_dinv<<<(NN + TPB - 1) / TPB, TPB, 0, stream>>>(deg, dinv);

  // stage 1: temporal attention
  k_rhs<<<nBNT / TPB, TPB, 0, stream>>>(X, U3, rhs);
  k_lhs0<<<(BB * TT * FIN) / TPB, TPB, 0, stream>>>(X, U1, lhs0);
  k_lhs<<<nBNT / TPB, TPB, 0, stream>>>(lhs0, U2, lhs);
  k_emat<<<BB, TPB, 0, stream>>>(lhs, rhs, be, Ve, EmatG);
  k_xt<<<nBNFT / TPB, TPB, 0, stream>>>(X, EmatG, Xt);

  // stage 2: spatial attention
  k_ls0<<<(BB * NN * FIN) / TPB, TPB, 0, stream>>>(Xt, W1, ls0);
  k_lsrs<<<nBNT / TPB, TPB, 0, stream>>>(Xt, ls0, W2, W3, lhs_s, rhsT);
  k_f32_to_f16<<<(NN * NN) / TPB, TPB, 0, stream>>>(Vs, Vs16, NN * NN);
  k_qst<<<(BB * NN * NN) / TPB, TPB, 0, stream>>>(lhs_s, rhsT, bs, QsT);
  // 2x4 register-blocked WMMA GEMM: BB*64*32 waves
  k_gemm_wmma<<<(BB * 64 * 32 * 32) / TPB, TPB, 0, stream>>>(QsT, Vs16, ST);
  k_softmax_rows<<<BB * NN, TPB, 0, stream>>>(ST);

  // stage 3: Chebyshev conv
  k_att<<<(BB * EE) / TPB, TPB, 0, stream>>>(ei, dinv, ST, att);
  k_diag_tx0<<<nBNFT / TPB, TPB, 0, stream>>>(X, ST, Tx0);
  k_zero<<<nBNFT / TPB, TPB, 0, stream>>>(Tx1, nBNFT);
  k_prop<<<(BB * EE * (FIN * TT)) / TPB, TPB, 0, stream>>>(ei, att, Tx0, Tx1);
  k_zero<<<nBNFT / TPB, TPB, 0, stream>>>(Tx2, nBNFT);
  k_prop<<<(BB * EE * (FIN * TT)) / TPB, TPB, 0, stream>>>(ei, att, Tx1, Tx2);
  k_tx2fix<<<nBNFT / TPB, TPB, 0, stream>>>(Tx2, Tx0, nBNFT);
  k_cheb_wmma<<<(BB * NN * 32) / TPB, TPB, 0, stream>>>(Tx0, Tx1, Tx2, chebW, chebB, Xhat);

  // stage 4: temporal conv + residual + LayerNorm -> d_out (B,N,TF,T)
  k_tconv_ln<<<(BB * NN * TT) / 4, TPB, 0, stream>>>(Xhat, X, Wt, bt, Wr, br,
                                                     gamma, beta, out);
  (void)in_sizes; (void)n_in; (void)out_size; (void)ws_size;
}
